// MetricalGNN_89936615178316
// MI455X (gfx1250) — compile-verified, matrix operand order checked by reading, not run on hardware
//
#include <hip/hip_runtime.h>
#include <hip/hip_bf16.h>
#include <math.h>

#define N_NODES 100000
#define EDGES   1600000
#define TYPES   3
#define DIM     128
#define NBATCH  64
#define GF_DIM  32
#define GO_DIM  64
#define KE_DIM  16
#define ZDIM    (DIM + GO_DIM + KE_DIM)   /* 208 */
#define OUT_DIM 11
#define LN_EPS  1e-5f
#define BN_EPS  1e-5f

typedef __attribute__((ext_vector_type(16))) __bf16         v16bf;
typedef __attribute__((ext_vector_type(8)))  float          v8f;
typedef __attribute__((ext_vector_type(8)))  unsigned short v8u;

union Frag32 { v8u u[2]; v16bf bf; };

__device__ __forceinline__ unsigned short f32_to_bf16(float f) {
  unsigned u = __float_as_uint(f);
  u += 0x7FFFu + ((u >> 16) & 1u);          // round-to-nearest-even
  return (unsigned short)(u >> 16);
}

__device__ __forceinline__ float wave_sum(float v) {
#pragma unroll
  for (int off = 16; off > 0; off >>= 1) v += __shfl_xor(v, off, 32);
  return v;
}

// ---------------------------------------------------------------------------
// WMMA GEMM, strip-mined: each wave computes a 16x128 strip of
// C[M,128] (f32) = A[M,128] (bf16) @ W (bf16, stored N-major WT[n][k]).
// The 32KB weight matrix is staged into LDS once per block with the CDNA5
// async global->LDS pipe (ASYNCcnt); B fragments are ds_load_b128, A
// fragments global_load_b128.  8 tiles/wave -> 32 WMMA/wave.
// FLAGS: bit0 = accumulate into C, bit1 = relu, bit2 = add bias
// block = 256 (8 waves), grid.x = ceil(M/16/8); M must be a multiple of 16.
// ---------------------------------------------------------------------------
template <int FLAGS>
__global__ __launch_bounds__(256)
void wmma_gemm_strip(const unsigned short* __restrict__ Abf,
                     const unsigned short* __restrict__ WTbf,
                     const float* __restrict__ bias,
                     float* __restrict__ C, int M) {
  __shared__ unsigned short Bsh[DIM * DIM];   // 32 KB of 320 KB/WGP

  const int tid = threadIdx.x;
  // async stage WT -> LDS: 256 threads x 8 x b128 = 32 KB, no VGPR round-trip
  {
    const unsigned lds0 = (unsigned)(size_t)&Bsh[0];   // LDS offset = addr[31:0]
#pragma unroll
    for (int i = 0; i < 8; ++i) {
      const int off = tid * 16 + i * 4096;             // bytes
      const char* gaddr = (const char*)WTbf + off;
      asm volatile("global_load_async_to_lds_b128 %0, %1, off"
                   :: "v"(lds0 + (unsigned)off), "v"(gaddr)
                   : "memory");
    }
    asm volatile("s_wait_asynccnt 0" ::: "memory");
  }
  __syncthreads();

  const int wave = tid >> 5;
  const int lane = tid & 31;
  const int wid  = blockIdx.x * 8 + wave;       // strip index
  const int row0 = wid * 16;
  if (row0 >= M) return;                        // wave-uniform: EXEC stays full

  const int half = lane >> 4;
  const int r    = lane & 15;
  const unsigned short* Arow = Abf + (size_t)(row0 + r) * DIM;

  v8f c[8];
#pragma unroll
  for (int n = 0; n < 8; ++n) c[n] = {};

#pragma unroll
  for (int kk = 0; kk < DIM; kk += 32) {
    Frag32 a;
    a.u[0] = *reinterpret_cast<const v8u*>(Arow + kk +      half * 8);
    a.u[1] = *reinterpret_cast<const v8u*>(Arow + kk + 16 + half * 8);
#pragma unroll
    for (int n = 0; n < 8; ++n) {
      Frag32 b;
      const unsigned short* Brow = &Bsh[(n * 16 + r) * DIM];
      b.u[0] = *reinterpret_cast<const v8u*>(Brow + kk +      half * 8);
      b.u[1] = *reinterpret_cast<const v8u*>(Brow + kk + 16 + half * 8);
      c[n] = __builtin_amdgcn_wmma_f32_16x16x32_bf16(false, a.bf, false, b.bf,
                                                     (short)0, c[n], false, false);
    }
  }

#pragma unroll
  for (int n = 0; n < 8; ++n) {
    float bn = (FLAGS & 4) ? bias[n * 16 + r] : 0.0f;
    float* Cbase = C + (size_t)(row0 + half * 8) * DIM + n * 16 + r;
#pragma unroll
    for (int v = 0; v < 8; ++v) {
      float* pp = Cbase + (size_t)v * DIM;
      float val = c[n][v] + bn;
      if (FLAGS & 1) val += *pp;
      if (FLAGS & 2) val = fmaxf(val, 0.0f);
      *pp = val;
    }
  }
}

// ---------------------------------------------------------------------------
// Utility / elementwise kernels
// ---------------------------------------------------------------------------
__global__ void zero_f32(float* __restrict__ p, size_t n) {
  size_t i = (size_t)blockIdx.x * blockDim.x + threadIdx.x;
  if (i < n) p[i] = 0.0f;
}

__global__ void cvt_bf16(const float* __restrict__ in, unsigned short* __restrict__ out, size_t n) {
  size_t i = (size_t)blockIdx.x * blockDim.x + threadIdx.x;
  if (i < n) out[i] = f32_to_bf16(in[i]);
}

// W (nmat x K x N, f32, row-major) -> WT (nmat x N x K, bf16)
__global__ void transpose_cvt_bf16(const float* __restrict__ W, unsigned short* __restrict__ WT, int nmat) {
  int idx   = blockIdx.x * blockDim.x + threadIdx.x;
  int total = nmat * DIM * DIM;
  if (idx >= total) return;
  int m   = idx >> 14;
  int rem = idx & 16383;
  int n   = rem >> 7;
  int k   = rem & 127;
  WT[idx] = f32_to_bf16(W[(size_t)m * DIM * DIM + (size_t)k * DIM + n]);
}

// scatter-add for segment mean: wave per edge, 4 floats per lane
__global__ __launch_bounds__(256)
void scatter_add(const float* __restrict__ feat, const int* __restrict__ src,
                 const int* __restrict__ dst, float* __restrict__ agg,
                 float* __restrict__ cnt, int nedges) {
  size_t gidx = (size_t)blockIdx.x * blockDim.x + threadIdx.x;
  int e    = (int)(gidx >> 5);
  int lane = (int)(gidx & 31);
  if (e >= nedges) return;
  int s = src[e], d = dst[e];
  float4 v = *reinterpret_cast<const float4*>(feat + (size_t)s * DIM + lane * 4);
  float* o = agg + (size_t)d * DIM + lane * 4;
  atomicAdd(o + 0, v.x); atomicAdd(o + 1, v.y);
  atomicAdd(o + 2, v.z); atomicAdd(o + 3, v.w);
  if (lane == 0) atomicAdd(cnt + d, 1.0f);
}

// agg_bf16 = (agg / max(cnt,1)) as bf16  (fused divide + convert)
__global__ void div_cvt(const float* __restrict__ agg, const float* __restrict__ cnt,
                        unsigned short* __restrict__ out, size_t total) {
  size_t i = (size_t)blockIdx.x * blockDim.x + threadIdx.x;
  if (i >= total) return;
  out[i] = f32_to_bf16(agg[i] / fmaxf(cnt[i >> 7], 1.0f));
}

// acc += o / max(||o||_2, 1e-12) rowwise  (wave per row)
__global__ __launch_bounds__(256)
void l2norm_add(const float* __restrict__ o, float* __restrict__ acc, int n) {
  int row = blockIdx.x * 8 + (threadIdx.x >> 5);
  if (row >= n) return;
  int lane = threadIdx.x & 31;
  float4 v = *reinterpret_cast<const float4*>(o + (size_t)row * DIM + lane * 4);
  float ss = wave_sum(v.x * v.x + v.y * v.y + v.z * v.z + v.w * v.w);
  float inv = 1.0f / fmaxf(sqrtf(ss), 1e-12f);
  float* a = acc + (size_t)row * DIM + lane * 4;
  float4 av = *reinterpret_cast<float4*>(a);
  av.x += v.x * inv; av.y += v.y * inv; av.z += v.z * inv; av.w += v.w * inv;
  *reinterpret_cast<float4*>(a) = av;
}

// out = mean over types (acc*inv_t), optional relu+LayerNorm; writes f32 and optional bf16
__global__ __launch_bounds__(256)
void mean_act_ln(const float* __restrict__ acc, float inv_t,
                 const float* __restrict__ g, const float* __restrict__ b,
                 float* __restrict__ outf, unsigned short* __restrict__ outbf,
                 int n, int do_ln) {
  int row = blockIdx.x * 8 + (threadIdx.x >> 5);
  if (row >= n) return;
  int lane = threadIdx.x & 31;
  int c0   = lane * 4;
  float4 v = *reinterpret_cast<const float4*>(acc + (size_t)row * DIM + c0);
  v.x *= inv_t; v.y *= inv_t; v.z *= inv_t; v.w *= inv_t;
  if (do_ln) {
    v.x = fmaxf(v.x, 0.f); v.y = fmaxf(v.y, 0.f);
    v.z = fmaxf(v.z, 0.f); v.w = fmaxf(v.w, 0.f);
    float mu = wave_sum(v.x + v.y + v.z + v.w) * (1.0f / DIM);
    float dx = v.x - mu, dy = v.y - mu, dz = v.z - mu, dw = v.w - mu;
    float var  = wave_sum(dx * dx + dy * dy + dz * dz + dw * dw) * (1.0f / DIM);
    float rstd = rsqrtf(var + LN_EPS);
    v.x = dx * rstd * g[c0 + 0] + b[c0 + 0];
    v.y = dy * rstd * g[c0 + 1] + b[c0 + 1];
    v.z = dz * rstd * g[c0 + 2] + b[c0 + 2];
    v.w = dw * rstd * g[c0 + 3] + b[c0 + 3];
  }
  *reinterpret_cast<float4*>(outf + (size_t)row * DIM + c0) = v;
  if (outbf) {
    unsigned short* ob = outbf + (size_t)row * DIM + c0;
    ob[0] = f32_to_bf16(v.x); ob[1] = f32_to_bf16(v.y);
    ob[2] = f32_to_bf16(v.z); ob[3] = f32_to_bf16(v.w);
  }
}

// ---------------------------------------------------------------------------
// Pooling
// ---------------------------------------------------------------------------
__global__ __launch_bounds__(256)
void gate_kernel(const float* __restrict__ note, const float* __restrict__ gw,
                 const float* __restrict__ gb, float* __restrict__ gate, int n) {
  int row = blockIdx.x * 8 + (threadIdx.x >> 5);
  if (row >= n) return;
  int lane = threadIdx.x & 31;
  float4 v = *reinterpret_cast<const float4*>(note + (size_t)row * DIM + lane * 4);
  float4 w = *reinterpret_cast<const float4*>(gw + lane * 4);
  float s = wave_sum(v.x * w.x + v.y * w.y + v.z * w.z + v.w * w.w) + gb[0];
  if (lane == 0) gate[row] = 1.0f / (1.0f + __expf(-s));
}

__global__ void gate_max(const float* __restrict__ gate, const int* __restrict__ batch,
                         unsigned* __restrict__ gmax_bits, int n) {
  int i = blockIdx.x * blockDim.x + threadIdx.x;
  if (i >= n) return;
  atomicMax(gmax_bits + batch[i], __float_as_uint(gate[i]));  // gate > 0
}

__global__ void gate_exp(const float* __restrict__ gate, const int* __restrict__ batch,
                         const unsigned* __restrict__ gmax_bits, float* __restrict__ ex,
                         float* __restrict__ den, int n) {
  int i = blockIdx.x * blockDim.x + threadIdx.x;
  if (i >= n) return;
  float e = __expf(gate[i] - __uint_as_float(gmax_bits[batch[i]]));
  ex[i] = e;
  atomicAdd(den + batch[i], e);
}

__global__ __launch_bounds__(256)
void pool_kernel(const float* __restrict__ note, const float* __restrict__ ex,
                 const float* __restrict__ den, const int* __restrict__ batch,
                 float* __restrict__ pooled, int n) {
  int row = blockIdx.x * 8 + (threadIdx.x >> 5);
  if (row >= n) return;
  int lane = threadIdx.x & 31;
  int bb   = batch[row];
  float a  = ex[row] / den[bb];
  float4 v = *reinterpret_cast<const float4*>(note + (size_t)row * DIM + lane * 4);
  float* o = pooled + (size_t)bb * DIM + lane * 4;
  atomicAdd(o + 0, a * v.x); atomicAdd(o + 1, a * v.y);
  atomicAdd(o + 2, a * v.z); atomicAdd(o + 3, a * v.w);
}

// ---------------------------------------------------------------------------
// Head (tiny)
// ---------------------------------------------------------------------------
__global__ void small_gemm(const float* __restrict__ A, const float* __restrict__ W,
                           const float* __restrict__ bias, float* __restrict__ C,
                           int M, int K, int Nc, int relu) {
  int idx = blockIdx.x * blockDim.x + threadIdx.x;
  if (idx >= M * Nc) return;
  int m = idx / Nc, n = idx % Nc;
  float acc = bias ? bias[n] : 0.0f;
  for (int k = 0; k < K; ++k) acc += A[(size_t)m * K + k] * W[(size_t)k * Nc + n];
  C[idx] = relu ? fmaxf(acc, 0.0f) : acc;
}

__global__ void build_z(const float* __restrict__ pooled, const float* __restrict__ gvec,
                        const float* __restrict__ kemb, const int* __restrict__ key,
                        float* __restrict__ z) {
  int idx = blockIdx.x * blockDim.x + threadIdx.x;
  if (idx >= NBATCH * ZDIM) return;
  int bi = idx / ZDIM, j = idx % ZDIM;
  float v;
  if (j < DIM)               v = pooled[bi * DIM + j];
  else if (j < DIM + GO_DIM) v = gvec[bi * GO_DIM + (j - DIM)];
  else                       v = kemb[key[bi] * KE_DIM + (j - DIM - GO_DIM)];
  z[idx] = v;
}

__global__ void bn_eval(float* __restrict__ z, const float* __restrict__ g,
                        const float* __restrict__ b, const float* __restrict__ rm,
                        const float* __restrict__ rv, int total, int C) {
  int idx = blockIdx.x * blockDim.x + threadIdx.x;
  if (idx >= total) return;
  int c = idx % C;
  z[idx] = (z[idx] - rm[c]) * rsqrtf(rv[c] + BN_EPS) * g[c] + b[c];
}

// ---------------------------------------------------------------------------
// Launch
// ---------------------------------------------------------------------------
extern "C" void kernel_launch(void* const* d_in, const int* in_sizes, int n_in,
                              void* d_out, int out_size, void* d_ws, size_t ws_size,
                              hipStream_t stream) {
  const float* x       = (const float*)d_in[0];
  const int*   eidx    = (const int*)d_in[1];
  const int*   batch   = (const int*)d_in[2];
  const int*   key     = (const int*)d_in[3];
  const float* gfeats  = (const float*)d_in[4];
  const float* proj_W  = (const float*)d_in[5];
  const float* proj_b  = (const float*)d_in[6];
  const float* l0_Wl   = (const float*)d_in[7];
  const float* l0_bl   = (const float*)d_in[8];
  const float* l0_Wr   = (const float*)d_in[9];
  const float* lW_l    = (const float*)d_in[10];
  const float* lb_l    = (const float*)d_in[11];
  const float* lW_r    = (const float*)d_in[12];
  const float* ln_g    = (const float*)d_in[13];
  const float* ln_b    = (const float*)d_in[14];
  const float* gate_W  = (const float*)d_in[15];
  const float* gate_b  = (const float*)d_in[16];
  const float* gfc_W   = (const float*)d_in[17];
  const float* gfc_b   = (const float*)d_in[18];
  const float* key_emb = (const float*)d_in[19];
  const float* m1_W    = (const float*)d_in[20];
  const float* m1_b    = (const float*)d_in[21];
  const float* bn_g    = (const float*)d_in[22];
  const float* bn_b    = (const float*)d_in[23];
  const float* bn_rm   = (const float*)d_in[24];
  const float* bn_rv   = (const float*)d_in[25];
  const float* m2_W    = (const float*)d_in[26];
  const float* m2_b    = (const float*)d_in[27];
  const float* m3_W    = (const float*)d_in[28];
  const float* m3_b    = (const float*)d_in[29];
  float*       out     = (float*)d_out;

  const size_t NH  = (size_t)N_NODES * DIM;
  const size_t MAT = (size_t)DIM * DIM;

  char* p = (char*)d_ws;
  auto carve = [&](size_t bytes) -> void* {
    void* r = (void*)p;
    p += (bytes + 255) & ~(size_t)255;
    return r;
  };
  float*          acc     = (float*)carve(NH * 4);
  float*          agg     = (float*)carve(NH * 4);
  float*          tmp     = (float*)carve(NH * 4);
  float*          hbuf    = (float*)carve(NH * 4);
  float*          cnt     = (float*)carve((size_t)N_NODES * 4);
  unsigned short* xbf     = (unsigned short*)carve(NH * 2);
  unsigned short* hbf     = (unsigned short*)carve(NH * 2);
  unsigned short* abf     = (unsigned short*)carve(NH * 2);
  unsigned short* wt_proj = (unsigned short*)carve(3 * MAT * 2);
  unsigned short* wt_l0wl = (unsigned short*)carve(3 * MAT * 2);
  unsigned short* wt_l0wr = (unsigned short*)carve(3 * MAT * 2);
  unsigned short* wt_lwl  = (unsigned short*)carve(6 * MAT * 2);
  unsigned short* wt_lwr  = (unsigned short*)carve(6 * MAT * 2);
  float*          gate    = (float*)carve((size_t)N_NODES * 4);
  float*          ex      = (float*)carve((size_t)N_NODES * 4);
  unsigned*       gmaxb   = (unsigned*)carve((size_t)NBATCH * 4);
  float*          den     = (float*)carve((size_t)NBATCH * 4);
  float*          pooled  = (float*)carve((size_t)NBATCH * DIM * 4);
  float*          gvec    = (float*)carve((size_t)NBATCH * GO_DIM * 4);
  float*          zbuf    = (float*)carve((size_t)NBATCH * ZDIM * 4);
  float*          z1      = (float*)carve((size_t)NBATCH * 256 * 4);
  float*          z2      = (float*)carve((size_t)NBATCH * 128 * 4);

  const int B256         = 256;
  const int blocks_NH    = (int)((NH + 255) / 256);
  const int blocks_rows  = (N_NODES + 7) / 8;            // wave-per-row kernels
  const int blocks_strip = (N_NODES / 16 + 7) / 8;       // 782: 8 strips/block
  const int blocks_edges = (int)(((size_t)EDGES * 32 + 255) / 256);
  const int blocks_N     = (N_NODES + 255) / 256;

  // ---- one-time conversions ----
  cvt_bf16<<<blocks_NH, B256, 0, stream>>>(x, xbf, NH);
  transpose_cvt_bf16<<<(3 * (int)MAT + 255) / 256, B256, 0, stream>>>(proj_W, wt_proj, 3);
  transpose_cvt_bf16<<<(3 * (int)MAT + 255) / 256, B256, 0, stream>>>(l0_Wl, wt_l0wl, 3);
  transpose_cvt_bf16<<<(3 * (int)MAT + 255) / 256, B256, 0, stream>>>(l0_Wr, wt_l0wr, 3);
  transpose_cvt_bf16<<<(6 * (int)MAT + 255) / 256, B256, 0, stream>>>(lW_l, wt_lwl, 6);
  transpose_cvt_bf16<<<(6 * (int)MAT + 255) / 256, B256, 0, stream>>>(lW_r, wt_lwr, 6);

  // ---- Layer 0: SAGE(project=True, normalize=True), mean over types ----
  zero_f32<<<blocks_NH, B256, 0, stream>>>(acc, NH);
  for (int t = 0; t < TYPES; ++t) {
    const int* src = eidx + ((size_t)t * 2 + 0) * EDGES;
    const int* dst = eidx + ((size_t)t * 2 + 1) * EDGES;
    // xp = relu(x @ proj_W[t] + proj_b[t])
    wmma_gemm_strip<6><<<blocks_strip, B256, 0, stream>>>(xbf, wt_proj + t * MAT,
                                                          proj_b + t * DIM, tmp, N_NODES);
    // agg = seg_mean(xp[src], dst)  (bf16 out, fused)
    zero_f32<<<blocks_NH, B256, 0, stream>>>(agg, NH);
    zero_f32<<<blocks_N, B256, 0, stream>>>(cnt, N_NODES);
    scatter_add<<<blocks_edges, B256, 0, stream>>>(tmp, src, dst, agg, cnt, EDGES);
    div_cvt<<<blocks_NH, B256, 0, stream>>>(agg, cnt, abf, NH);
    // o = agg @ Wl + bl + x @ Wr
    wmma_gemm_strip<4><<<blocks_strip, B256, 0, stream>>>(abf, wt_l0wl + t * MAT,
                                                          l0_bl + t * DIM, tmp, N_NODES);
    wmma_gemm_strip<1><<<blocks_strip, B256, 0, stream>>>(xbf, wt_l0wr + t * MAT,
                                                          nullptr, tmp, N_NODES);
    // acc += F.normalize(o)
    l2norm_add<<<blocks_rows, B256, 0, stream>>>(tmp, acc, N_NODES);
  }
  // h = LN(relu(mean)); also bf16 copy
  mean_act_ln<<<blocks_rows, B256, 0, stream>>>(acc, 1.0f / TYPES, ln_g, ln_b,
                                                hbuf, hbf, N_NODES, 1);

  // ---- Layers 1..2: plain SAGE, mean over types ----
  for (int li = 0; li < 2; ++li) {
    zero_f32<<<blocks_NH, B256, 0, stream>>>(acc, NH);
    for (int t = 0; t < TYPES; ++t) {
      const int* src = eidx + ((size_t)t * 2 + 0) * EDGES;
      const int* dst = eidx + ((size_t)t * 2 + 1) * EDGES;
      zero_f32<<<blocks_NH, B256, 0, stream>>>(agg, NH);
      zero_f32<<<blocks_N, B256, 0, stream>>>(cnt, N_NODES);
      scatter_add<<<blocks_edges, B256, 0, stream>>>(hbuf, src, dst, agg, cnt, EDGES);
      div_cvt<<<blocks_NH, B256, 0, stream>>>(agg, cnt, abf, NH);
      // acc += agg @ Wl + bl + h @ Wr
      wmma_gemm_strip<5><<<blocks_strip, B256, 0, stream>>>(abf, wt_lwl + (li * 3 + t) * MAT,
                                                            lb_l + (li * 3 + t) * DIM,
                                                            acc, N_NODES);
      wmma_gemm_strip<1><<<blocks_strip, B256, 0, stream>>>(hbf, wt_lwr + (li * 3 + t) * MAT,
                                                            nullptr, acc, N_NODES);
    }
    if (li == 0) {
      mean_act_ln<<<blocks_rows, B256, 0, stream>>>(acc, 1.0f / TYPES, ln_g + DIM, ln_b + DIM,
                                                    hbuf, hbf, N_NODES, 1);
    } else {
      mean_act_ln<<<blocks_rows, B256, 0, stream>>>(acc, 1.0f / TYPES, nullptr, nullptr,
                                                    hbuf, nullptr, N_NODES, 0);  // note
    }
  }

  // ---- attentional pooling ----
  gate_kernel<<<blocks_rows, B256, 0, stream>>>(hbuf, gate_W, gate_b, gate, N_NODES);
  zero_f32<<<1, B256, 0, stream>>>((float*)gmaxb, NBATCH);      // 0 < all sigmoid gates
  zero_f32<<<1, B256, 0, stream>>>(den, NBATCH);
  zero_f32<<<(NBATCH * DIM + 255) / 256, B256, 0, stream>>>(pooled, (size_t)NBATCH * DIM);
  gate_max<<<blocks_N, B256, 0, stream>>>(gate, batch, gmaxb, N_NODES);
  gate_exp<<<blocks_N, B256, 0, stream>>>(gate, batch, gmaxb, ex, den, N_NODES);
  pool_kernel<<<blocks_rows, B256, 0, stream>>>(hbuf, ex, den, batch, pooled, N_NODES);

  // ---- head ----
  small_gemm<<<(NBATCH * GO_DIM + 255) / 256, B256, 0, stream>>>(gfeats, gfc_W, gfc_b, gvec,
                                                                 NBATCH, GF_DIM, GO_DIM, 1);
  build_z<<<(NBATCH * ZDIM + 255) / 256, B256, 0, stream>>>(pooled, gvec, key_emb, key, zbuf);
  small_gemm<<<(NBATCH * 256 + 255) / 256, B256, 0, stream>>>(zbuf, m1_W, m1_b, z1,
                                                              NBATCH, ZDIM, 256, 1);
  bn_eval<<<(NBATCH * 256 + 255) / 256, B256, 0, stream>>>(z1, bn_g, bn_b, bn_rm, bn_rv,
                                                           NBATCH * 256, 256);
  small_gemm<<<(NBATCH * 128 + 255) / 256, B256, 0, stream>>>(z1, m2_W, m2_b, z2,
                                                              NBATCH, 256, 128, 1);
  small_gemm<<<(NBATCH * OUT_DIM + 255) / 256, B256, 0, stream>>>(z2, m3_W, m3_b, out,
                                                                  NBATCH, 128, OUT_DIM, 0);
  (void)in_sizes; (void)n_in; (void)out_size; (void)ws_size;
}
